// GAT_65867618451460
// MI455X (gfx1250) — matrix operand order repaired and checked by reading, not compiled, not run on hardware
//
#include <hip/hip_runtime.h>
#include <hip/hip_bf16.h>

#define N_NODES 10000
#define N_EDGES 320000
#define HID 256
#define EDIM 16
#define NGRAPH 64

typedef __attribute__((ext_vector_type(16))) __bf16 v16bf;
typedef __attribute__((ext_vector_type(8)))  float  v8f;

union FragBF { v16bf v; unsigned u[8]; };

// ---------------- conversions ----------------
__global__ void cvt_bf16_k(const float* __restrict__ in, __hip_bfloat16* __restrict__ out, int n) {
    int i = blockIdx.x * 256 + threadIdx.x;
    if (i < n) out[i] = __float2bfloat16(in[i]);
}

// W [K,256] fp32 -> Wt [256,K] bf16 (transposed so B-fragment K-pairs are contiguous)
__global__ void cvt_wt_k(const float* __restrict__ W, __hip_bfloat16* __restrict__ Wt, int K) {
    int i = blockIdx.x * 256 + threadIdx.x;
    if (i < K * HID) {
        int k = i >> 8, n = i & 255;
        Wt[n * K + k] = __float2bfloat16(W[i]);
    }
}

// ---------------- dual GEMM: xl = A@Wl + bl, xr = A@Wr + br ----------------
// A bf16 [10000,K] row-major; Wlt/Wrt bf16 [256,K] (transposed).
// One wave computes a 32x64 macro-tile for BOTH weight matrices:
//   2 M-tiles x 4 N-tiles x {Wl,Wr} = 16 WMMA accumulators (128 VGPRs).
// A frags reused across 4 N-tiles, B frags across 2 M-tiles -> 16 wmma per K-step.
__global__ void __launch_bounds__(32)
gemm_dual_k(const __hip_bfloat16* __restrict__ A,
            const __hip_bfloat16* __restrict__ Wlt,
            const __hip_bfloat16* __restrict__ Wrt,
            const float* __restrict__ bl, const float* __restrict__ br,
            float* __restrict__ xl, float* __restrict__ xr, int K) {
    int lane = threadIdx.x & 31;
    int half = lane >> 4;
    int l15  = lane & 15;
    int tm  = blockIdx.x * 32;   // 2 M-tiles
    int tn0 = blockIdx.y * 64;   // 4 N-tiles

    v8f cl[2][4], cr[2][4];
    #pragma unroll
    for (int mt = 0; mt < 2; ++mt)
        #pragma unroll
        for (int nt = 0; nt < 4; ++nt) {
            cl[mt][nt] = v8f{0.f,0.f,0.f,0.f,0.f,0.f,0.f,0.f};
            cr[mt][nt] = v8f{0.f,0.f,0.f,0.f,0.f,0.f,0.f,0.f};
        }

    // A row pointers (clamped: last block's second M-tile reads a dup row; stores guarded)
    const unsigned* arow[2];
    #pragma unroll
    for (int mt = 0; mt < 2; ++mt) {
        int row = tm + mt * 16 + l15;
        row = row < N_NODES ? row : (N_NODES - 1);
        arow[mt] = reinterpret_cast<const unsigned*>(A + (size_t)row * K);
    }
    const unsigned* wlr[4];
    const unsigned* wrr[4];
    #pragma unroll
    for (int nt = 0; nt < 4; ++nt) {
        int n = tn0 + nt * 16 + l15;
        wlr[nt] = reinterpret_cast<const unsigned*>(Wlt + (size_t)n * K);
        wrr[nt] = reinterpret_cast<const unsigned*>(Wrt + (size_t)n * K);
    }

    for (int kb = 0; kb < K; kb += 32) {
        int kb2 = kb >> 1; // dword (bf16-pair) index
        FragBF a[2];
        #pragma unroll
        for (int mt = 0; mt < 2; ++mt)
            #pragma unroll
            for (int v = 0; v < 8; ++v) {
                // A 16x32 layout: lanes0-15 M=0..15; VGPR0-3: K=half*8+2v ; VGPR4-7: +16
                int adw = ((v < 4) ? 0 : 8) + half * 4 + (v & 3);
                a[mt].u[v] = arow[mt][kb2 + adw];
            }
        #pragma unroll
        for (int nt = 0; nt < 4; ++nt) {
            FragBF fl, fr;
            #pragma unroll
            for (int v = 0; v < 8; ++v) {
                // B 32x16 layout: same K order as A; VGPR0-3: K=half*8+2v ; VGPR4-7: +16
                int bdw = ((v < 4) ? 0 : 8) + half * 4 + (v & 3);
                fl.u[v] = wlr[nt][kb2 + bdw];
                fr.u[v] = wrr[nt][kb2 + bdw];
            }
            #pragma unroll
            for (int mt = 0; mt < 2; ++mt) {
                cl[mt][nt] = __builtin_amdgcn_wmma_f32_16x16x32_bf16(false, a[mt].v, false, fl.v, (short)0, cl[mt][nt], false, false);
                cr[mt][nt] = __builtin_amdgcn_wmma_f32_16x16x32_bf16(false, a[mt].v, false, fr.v, (short)0, cr[mt][nt], false, false);
            }
        }
    }

    #pragma unroll
    for (int nt = 0; nt < 4; ++nt) {
        int n = tn0 + nt * 16 + l15;
        float vbl = bl[n], vbr = br[n];
        #pragma unroll
        for (int mt = 0; mt < 2; ++mt)
            #pragma unroll
            for (int r = 0; r < 8; ++r) {
                int row = tm + mt * 16 + r + half * 8; // C/D layout: lanes16-31 hold M=8..15
                if (row < N_NODES) {
                    xl[(size_t)row * HID + n] = cl[mt][nt][r] + vbl;
                    xr[(size_t)row * HID + n] = cr[mt][nt][r] + vbr;
                }
            }
    }
}

// ---------------- per-layer init ----------------
__global__ void layer_init_k(float* __restrict__ num, float* __restrict__ den, int* __restrict__ smax) {
    int i = blockIdx.x * 256 + threadIdx.x;   // grid = N_NODES blocks => N_NODES*HID threads
    num[i] = 0.f;
    if (i < N_NODES) { den[i] = 0.f; smax[i] = (int)0x80000000; }
}

// ---------------- edge scores: s = att . leakyrelu(xl[src]+xr[dst]+ea@We, 0.2) ----------------
// one wave per edge; We (16KB) + att in LDS; segment max via mapped-int atomicMax.
__global__ void __launch_bounds__(256)
edge_scores_k(const int* __restrict__ src, const int* __restrict__ dst,
              const float* __restrict__ ea, const float* __restrict__ We,
              const float* __restrict__ att,
              const float* __restrict__ xl, const float* __restrict__ xr,
              float* __restrict__ s_out, int* __restrict__ smax, int E) {
    __shared__ float sWe[EDIM * HID];
    __shared__ float sAtt[HID];
    int tid = threadIdx.x;
    for (int i = tid; i < EDIM * HID; i += 256) sWe[i] = We[i];
    sAtt[tid] = att[tid];
    __syncthreads();

    int wid = tid >> 5, lane = tid & 31;
    int e = blockIdx.x * 8 + wid;
    if (e >= E) return;

    int sv = src[e], dv = dst[e];
    float a[EDIM];
    const float* eap = ea + (size_t)e * EDIM;
    #pragma unroll
    for (int d = 0; d < EDIM; ++d) a[d] = eap[d];   // broadcast across wave (same line)

    const float* xlp = xl + (size_t)sv * HID;
    const float* xrp = xr + (size_t)dv * HID;
    float part = 0.f;
    #pragma unroll
    for (int jj = 0; jj < 8; ++jj) {
        int j = jj * 32 + lane;                      // conflict-free LDS, coalesced global
        float ev = 0.f;
        #pragma unroll
        for (int d = 0; d < EDIM; ++d) ev += a[d] * sWe[d * HID + j];
        float m = xlp[j] + xrp[j] + ev;
        m = m > 0.f ? m : 0.2f * m;                  // leaky_relu(0.2)
        part += m * sAtt[j];
    }
    #pragma unroll
    for (int off = 16; off > 0; off >>= 1) part += __shfl_down(part, off, 32);

    if (lane == 0) {
        s_out[e] = part;
        int mi = __float_as_int(part);
        mi = mi >= 0 ? mi : (mi ^ 0x7fffffff);       // order-preserving int map
        atomicMax(&smax[dv], mi);
    }
}

// ---------------- edge scatter: num[dst] += ex * xl[src]; den[dst] += ex ----------------
__global__ void __launch_bounds__(256)
edge_scatter_k(const int* __restrict__ src, const int* __restrict__ dst,
               const float* __restrict__ s_in, const int* __restrict__ smax,
               const float* __restrict__ xl,
               float* __restrict__ num, float* __restrict__ den, int E) {
    int tid = threadIdx.x;
    int wid = tid >> 5, lane = tid & 31;
    int e = blockIdx.x * 8 + wid;
    if (e >= E) return;

    int sv = src[e], dv = dst[e];
    int mi = smax[dv];
    mi = mi >= 0 ? mi : (mi ^ 0x7fffffff);           // unmap (involution)
    float mx = __int_as_float(mi);
    float ex = __expf(s_in[e] - mx);

    const float* xlp = xl + (size_t)sv * HID;
    float* np = num + (size_t)dv * HID;
    #pragma unroll
    for (int jj = 0; jj < 8; ++jj) {
        int j = jj * 32 + lane;
        atomicAdd(&np[j], ex * xlp[j]);              // L2-resident f32 atomics
    }
    if (lane == 0) atomicAdd(&den[dv], ex);
}

// ---------------- normalize + bias + BN(eval) + activation; emits fp32 h and bf16 A ----------------
__global__ void node_norm_k(const float* __restrict__ num, const float* __restrict__ den,
                            const float* __restrict__ bias,
                            const float* __restrict__ gamma, const float* __restrict__ beta,
                            const float* __restrict__ mean, const float* __restrict__ var,
                            float* __restrict__ h, __hip_bfloat16* __restrict__ abf, int use_relu) {
    int idx = blockIdx.x * 256 + threadIdx.x;        // grid = N_NODES => N_NODES*HID threads
    int n = idx >> 8, j = idx & 255;
    float d = den[n];
    float v = d > 0.f ? num[idx] / d : 0.f;          // (sum ex*xl)/(sum ex); empty segment -> 0
    v += bias[j];
    v = (v - mean[j]) * rsqrtf(var[j] + 1e-5f) * gamma[j] + beta[j];
    v = v > 0.f ? v : (use_relu ? 0.f : 0.01f * v);
    h[idx] = v;
    abf[idx] = __float2bfloat16(v);
}

// ---------------- global mean pool + two linear heads ----------------
__global__ void pool_init_k(float* __restrict__ gsum, float* __restrict__ gcnt) {
    int i = blockIdx.x * 256 + threadIdx.x;          // grid = 64 => NGRAPH*HID threads
    gsum[i] = 0.f;
    if (i < NGRAPH) gcnt[i] = 0.f;
}

__global__ void pool_scatter_k(const float* __restrict__ h, const int* __restrict__ batch,
                               float* __restrict__ gsum, float* __restrict__ gcnt) {
    int idx = blockIdx.x * 256 + threadIdx.x;
    int n = idx >> 8, j = idx & 255;
    int b = batch[n];
    atomicAdd(&gsum[(size_t)b * HID + j], h[idx]);
    if (j == 0) atomicAdd(&gcnt[b], 1.0f);
}

__global__ void head_k(const float* __restrict__ gsum, const float* __restrict__ gcnt,
                       const float* __restrict__ w1, const float* __restrict__ b1,
                       const float* __restrict__ w2, const float* __restrict__ b2,
                       float* __restrict__ out) {
    int t = blockIdx.x * 256 + threadIdx.x;          // 1024 outputs: [pred1(64x8), pred2(64x8)]
    if (t >= 2 * NGRAPH * 8) return;
    int which = t >> 9;
    int g = (t >> 3) & 63;
    int o = t & 7;
    const float* wm = which ? w2 : w1;
    const float* bb = which ? b2 : b1;
    float cnt = gcnt[g]; cnt = cnt > 1.f ? cnt : 1.f;
    float acc = 0.f;
    for (int k = 0; k < HID; ++k) acc += (gsum[g * HID + k] / cnt) * wm[k * 8 + o];
    out[which * (NGRAPH * 8) + g * 8 + o] = acc + bb[o];
}

// ---------------- host launcher ----------------
extern "C" void kernel_launch(void* const* d_in, const int* in_sizes, int n_in,
                              void* d_out, int out_size, void* d_ws, size_t ws_size,
                              hipStream_t stream) {
    const float* x          = (const float*)d_in[0];
    const int*   edge_index = (const int*)  d_in[1];
    const float* edge_attr  = (const float*)d_in[2];
    const int*   batch      = (const int*)  d_in[3];
    const float* Wl1 = (const float*)d_in[4];
    const float* Wr1 = (const float*)d_in[5];
    const float* We1 = (const float*)d_in[6];
    const float* bl1 = (const float*)d_in[7];
    const float* br1 = (const float*)d_in[8];
    const float* att1= (const float*)d_in[9];
    const float* b1  = (const float*)d_in[10];
    const float* Wls = (const float*)d_in[11];
    const float* Wrs = (const float*)d_in[12];
    const float* Wes = (const float*)d_in[13];
    const float* bls = (const float*)d_in[14];
    const float* brs = (const float*)d_in[15];
    const float* atts= (const float*)d_in[16];
    const float* bs  = (const float*)d_in[17];
    const float* bn_gamma = (const float*)d_in[18];
    const float* bn_beta  = (const float*)d_in[19];
    const float* bn_mean  = (const float*)d_in[20];
    const float* bn_var   = (const float*)d_in[21];
    const float* lin1_w = (const float*)d_in[22];
    const float* lin1_b = (const float*)d_in[23];
    const float* lin2_w = (const float*)d_in[24];
    const float* lin2_b = (const float*)d_in[25];

    const int* src = edge_index;
    const int* dst = edge_index + N_EDGES;

    // workspace carve-up (all 256B aligned)
    char* w = (char*)d_ws;
    auto carve = [&](size_t bytes) { void* p = (void*)w; w += (bytes + 255) & ~(size_t)255; return p; };
    __hip_bfloat16* Abf = (__hip_bfloat16*)carve((size_t)N_NODES * HID * 2);
    __hip_bfloat16* Wlt = (__hip_bfloat16*)carve((size_t)HID * HID * 2);
    __hip_bfloat16* Wrt = (__hip_bfloat16*)carve((size_t)HID * HID * 2);
    float* xl   = (float*)carve((size_t)N_NODES * HID * 4);
    float* xr   = (float*)carve((size_t)N_NODES * HID * 4);
    float* num  = (float*)carve((size_t)N_NODES * HID * 4);
    float* h    = (float*)carve((size_t)N_NODES * HID * 4);
    float* den  = (float*)carve((size_t)N_NODES * 4);
    int*   smax = (int*)  carve((size_t)N_NODES * 4);
    float* sbuf = (float*)carve((size_t)N_EDGES * 4);
    float* gsum = (float*)carve((size_t)NGRAPH * HID * 4);
    float* gcnt = (float*)carve((size_t)NGRAPH * 4);

    for (int L = 0; L < 5; ++L) {
        int K = (L == 0) ? 128 : HID;
        const float* Wl  = (L == 0) ? Wl1  : Wls  + (size_t)(L - 1) * HID * HID;
        const float* Wr  = (L == 0) ? Wr1  : Wrs  + (size_t)(L - 1) * HID * HID;
        const float* We  = (L == 0) ? We1  : Wes  + (size_t)(L - 1) * EDIM * HID;
        const float* blv = (L == 0) ? bl1  : bls  + (size_t)(L - 1) * HID;
        const float* brv = (L == 0) ? br1  : brs  + (size_t)(L - 1) * HID;
        const float* att = (L == 0) ? att1 : atts + (size_t)(L - 1) * HID;
        const float* bb  = (L == 0) ? b1   : bs   + (size_t)(L - 1) * HID;

        if (L == 0)
            cvt_bf16_k<<<(N_NODES * 128 + 255) / 256, 256, 0, stream>>>(x, Abf, N_NODES * 128);
        cvt_wt_k<<<(K * HID + 255) / 256, 256, 0, stream>>>(Wl, Wlt, K);
        cvt_wt_k<<<(K * HID + 255) / 256, 256, 0, stream>>>(Wr, Wrt, K);

        // 32-row M macro-tiles: ceil(10000/32)=313 ; 4 N macro-tiles of 64 cols
        gemm_dual_k<<<dim3((N_NODES + 31) / 32, HID / 64), 32, 0, stream>>>(Abf, Wlt, Wrt, blv, brv, xl, xr, K);

        layer_init_k<<<N_NODES, 256, 0, stream>>>(num, den, smax);
        edge_scores_k<<<N_EDGES / 8, 256, 0, stream>>>(src, dst, edge_attr, We, att, xl, xr, sbuf, smax, N_EDGES);
        edge_scatter_k<<<N_EDGES / 8, 256, 0, stream>>>(src, dst, sbuf, smax, xl, num, den, N_EDGES);
        node_norm_k<<<N_NODES, 256, 0, stream>>>(num, den, bb,
                                                 bn_gamma + (size_t)L * HID, bn_beta + (size_t)L * HID,
                                                 bn_mean + (size_t)L * HID,  bn_var + (size_t)L * HID,
                                                 h, Abf, (L == 4) ? 1 : 0);
    }

    pool_init_k<<<NGRAPH, 256, 0, stream>>>(gsum, gcnt);
    pool_scatter_k<<<N_NODES, 256, 0, stream>>>(h, batch, gsum, gcnt);
    head_k<<<4, 256, 0, stream>>>(gsum, gcnt, lin1_w, lin1_b, lin2_w, lin2_b, (float*)d_out);
}
